// SelfAttention_56238301773977
// MI455X (gfx1250) — matrix operand order repaired and checked by reading, not compiled, not run on hardware
//
#include <hip/hip_runtime.h>

// ---------------------------------------------------------------------------
// MI455X (gfx1250) self-attention. Compute-bound (~4.1e11 FLOP vs ~1.5GB of
// HBM traffic @ 23.3 TB/s), so all four GEMMs run on v_wmma_f32_16x16x32_bf16
// with fp32 accumulation. Global->LDS uses CDNA5 GLOBAL_LOAD_ASYNC_TO_LDS_B128
// (ASYNCcnt) with double-buffered LDS and one barrier per K-step.
// ---------------------------------------------------------------------------

typedef __attribute__((ext_vector_type(16))) __bf16 v16bf;
typedef __attribute__((ext_vector_type(8)))  __bf16 v8bf;
typedef __attribute__((ext_vector_type(8)))  float  v8f;

#define LDK 40   // padded LDS row stride (elements): 80B rows -> conflict-free b128

static constexpr int TSEQ  = 2048;
static constexpr int CDIM  = 512;
static constexpr int HEADS = 4;
static constexpr int BATCH = 8;
static constexpr int BCH   = 2;   // batches per attention chunk (bounds ws use)
static constexpr float ATT_SCALE = 0.08838834764831845f; // (512/4)^-0.5

__device__ __forceinline__ v8f zero8() {
  v8f z;
#pragma unroll
  for (int i = 0; i < 8; ++i) z[i] = 0.0f;
  return z;
}

// ---- CDNA5 async global->LDS copies (ASYNCcnt) ----------------------------
// inst offset:imm is added to BOTH the LDS and the global address, so one
// (ldsaddr, gaddr) pair covers a contiguous 32B/64B chunk in both spaces.
__device__ __forceinline__ void async_ld_32B(unsigned lds, const __bf16* g) {
  unsigned long long ga = (unsigned long long)(size_t)g;
  asm volatile(
      "global_load_async_to_lds_b128 %0, %1, off\n\t"
      "global_load_async_to_lds_b128 %0, %1, off offset:16"
      :: "v"(lds), "v"(ga) : "memory");
}
__device__ __forceinline__ void async_ld_64B(unsigned lds, const __bf16* g) {
  unsigned long long ga = (unsigned long long)(size_t)g;
  asm volatile(
      "global_load_async_to_lds_b128 %0, %1, off\n\t"
      "global_load_async_to_lds_b128 %0, %1, off offset:16\n\t"
      "global_load_async_to_lds_b128 %0, %1, off offset:32\n\t"
      "global_load_async_to_lds_b128 %0, %1, off offset:48"
      :: "v"(lds), "v"(ga) : "memory");
}
__device__ __forceinline__ void wait_async0() {
  asm volatile("s_wait_asynccnt 0x0" ::: "memory");
}

// ---- WMMA fragment loads from LDS (wave32) --------------------------------
// A (16x32 MxK): lanes 0-15 row=lane, K[0..7]=V0-3, K[16..23]=V4-7;
//                lanes 16-31 row=lane-16, K[8..15], K[24..31].
__device__ __forceinline__ v16bf ldsA_frag(const __bf16* tile) {
  int lane = threadIdx.x & 31;
  const __bf16* p = tile + (lane & 15) * LDK + ((lane >> 4) << 3);
  union { v16bf v; v8bf h[2]; } u;
  u.h[0] = *(const v8bf*)p;
  u.h[1] = *(const v8bf*)(p + 16);
  return u.v;
}
// B (32x16 KxN) stored in LDS as [N][K]: lanes 0-15 col=lane hold K[0..7],
// K[16..23]; lanes 16-31 col=lane-16 hold K[8..15], K[24..31].
__device__ __forceinline__ v16bf ldsB_frag(const __bf16* tile) {
  int lane = threadIdx.x & 31;
  const __bf16* p = tile + (lane & 15) * LDK + ((lane >> 4) << 3);
  union { v16bf v; v8bf h[2]; } u;
  u.h[0] = *(const v8bf*)p;
  u.h[1] = *(const v8bf*)(p + 16);
  return u.v;
}

// ---- 128x256 GEMM core: C = A[M,K] * B[N,K]^T, fp32 accum -----------------
// 8 waves in 2(M)x4(N); each wave: 64x64 = 4x4 WMMA tiles. Double-buffered
// LDS fed by async copies; one barrier per 32-wide K step.
__device__ __forceinline__ void gemm_core(const __bf16* A, int lda,
                                          const __bf16* B, int ldb,
                                          int K, v8f acc[4][4]) {
  __shared__ __bf16 shA[2][128 * LDK];
  __shared__ __bf16 shB[2][256 * LDK];
  int tid = threadIdx.x;
  // A fill: 2 threads/row, 32B each.  B fill: 1 thread/row, 64B each.
  int arow = tid >> 1;
  int ac0  = (tid & 1) << 4;
  const __bf16* ga = A + (size_t)arow * lda + ac0;
  const __bf16* gb = B + (size_t)tid * ldb;
  unsigned ldsA0 = (unsigned)(size_t)&shA[0][arow * LDK + ac0];
  unsigned ldsA1 = (unsigned)(size_t)&shA[1][arow * LDK + ac0];
  unsigned ldsB0 = (unsigned)(size_t)&shB[0][tid * LDK];
  unsigned ldsB1 = (unsigned)(size_t)&shB[1][tid * LDK];

  int wid = tid >> 5;
  int aBand = (wid >> 2) * 64 * LDK;
  int bBand = (wid & 3) * 64 * LDK;

  // prologue: stage tile 0 into buffer 0
  async_ld_32B(ldsA0, ga);
  async_ld_64B(ldsB0, gb);
  ga += 32; gb += 32;

  for (int k0 = 0, p = 0; k0 < K; k0 += 32, p ^= 1) {
    wait_async0();       // own async copies for buf[p] have landed
    __syncthreads();     // everyone's copies landed; everyone done reading p^1
    if (k0 + 32 < K) {   // stage tile k+1 into buf[p^1] (overlaps compute)
      async_ld_32B(p ? ldsA0 : ldsA1, ga);
      async_ld_64B(p ? ldsB0 : ldsB1, gb);
      ga += 32; gb += 32;
    }
    const __bf16* aB = &shA[p][aBand];
    const __bf16* bB = &shB[p][bBand];
    v16bf af[4], bfv[4];
#pragma unroll
    for (int mi = 0; mi < 4; ++mi) af[mi] = ldsA_frag(aB + mi * 16 * LDK);
#pragma unroll
    for (int ni = 0; ni < 4; ++ni) bfv[ni] = ldsB_frag(bB + ni * 16 * LDK);
#pragma unroll
    for (int mi = 0; mi < 4; ++mi)
#pragma unroll
      for (int ni = 0; ni < 4; ++ni)
        acc[mi][ni] = __builtin_amdgcn_wmma_f32_16x16x32_bf16(
            false, af[mi], false, bfv[ni], (short)0, acc[mi][ni], false, false);
  }
}

#define ZERO_ACC                                               \
  v8f acc[4][4];                                               \
  _Pragma("unroll") for (int mi = 0; mi < 4; ++mi)             \
      _Pragma("unroll") for (int ni = 0; ni < 4; ++ni)         \
          acc[mi][ni] = zero8();

// D-tile per-lane coords: VGPR r -> M = r + 8*(lane>>4), N = lane&15.
#define EPILOGUE_COORDS                                        \
  int lane  = threadIdx.x & 31;                                \
  int wid   = threadIdx.x >> 5;                                \
  int mBase = (wid >> 2) * 64 + ((lane >> 4) << 3);            \
  int nBase = (wid & 3) * 64 + (lane & 15);

// ---- fp32 -> bf16 bulk conversion (one-shot, feeds all GEMM operands) -----
__global__ __launch_bounds__(256) void cvt_kernel(const float* __restrict__ in,
                                                  __bf16* __restrict__ out) {
  size_t i = ((size_t)blockIdx.x * 256 + threadIdx.x) * 8;
  const float4* s = (const float4*)(in + i);
  float4 f0 = s[0], f1 = s[1];
  v8bf o;
  o[0] = (__bf16)f0.x; o[1] = (__bf16)f0.y; o[2] = (__bf16)f0.z; o[3] = (__bf16)f0.w;
  o[4] = (__bf16)f1.x; o[5] = (__bf16)f1.y; o[6] = (__bf16)f1.z; o[7] = (__bf16)f1.w;
  *(v8bf*)(out + i) = o;
}

// ---- Kernel 1: fused QKV projection, scatter to Q/K/[V^T] bf16 ------------
__global__ __launch_bounds__(256) void qkv_gemm_kernel(
    const __bf16* __restrict__ xb, const __bf16* __restrict__ wb,
    __bf16* __restrict__ q, __bf16* __restrict__ kmat,
    __bf16* __restrict__ vT) {
  int m0 = blockIdx.x * 128;
  int n0 = blockIdx.y * 256;
  ZERO_ACC
  gemm_core(xb + (size_t)m0 * CDIM, CDIM, wb + (size_t)n0 * CDIM, CDIM, CDIM,
            acc);
  EPILOGUE_COORDS
  int three = n0 >> 11;          // 0=q 1=k 2=v (256-tile never straddles)
  int h     = (n0 >> 9) & 3;
  int cb    = n0 & (CDIM - 1);
  __bf16* dst01 = (three == 0) ? q : kmat;
#pragma unroll
  for (int mi = 0; mi < 4; ++mi)
#pragma unroll
    for (int ni = 0; ni < 4; ++ni)
#pragma unroll
      for (int r = 0; r < 8; ++r) {
        int m = m0 + mBase + mi * 16 + r;
        int c = cb + nBase + ni * 16;
        int b = m >> 11, t = m & (TSEQ - 1);
        size_t bh = (size_t)(b * HEADS + h);
        __bf16 val = (__bf16)acc[mi][ni][r];
        if (three < 2)
          dst01[(bh * TSEQ + t) * CDIM + c] = val;       // [B,H,T,C]
        else
          vT[(bh * CDIM + c) * TSEQ + t] = val;          // [B,H,C,T]
      }
}

// ---- Kernel 2: S = scale * Q K^T (fp32) -----------------------------------
__global__ __launch_bounds__(256) void scores_kernel(
    const __bf16* __restrict__ q, const __bf16* __restrict__ kmat,
    float* __restrict__ S, int b0) {
  int q0 = blockIdx.x * 128;
  int n0 = blockIdx.y * 256;
  int bc = blockIdx.z >> 2, h = blockIdx.z & 3;
  size_t bh = (size_t)((b0 + bc) * HEADS + h);
  ZERO_ACC
  gemm_core(q + (bh * TSEQ + q0) * CDIM, CDIM,
            kmat + (bh * TSEQ + n0) * CDIM, CDIM, CDIM, acc);
  EPILOGUE_COORDS
  float* out = S + ((size_t)(bc * HEADS + h) * TSEQ + q0) * TSEQ + n0;
#pragma unroll
  for (int mi = 0; mi < 4; ++mi)
#pragma unroll
    for (int ni = 0; ni < 4; ++ni)
#pragma unroll
      for (int r = 0; r < 8; ++r)
        out[(size_t)(mBase + mi * 16 + r) * TSEQ + nBase + ni * 16] =
            acc[mi][ni][r] * ATT_SCALE;
}

// ---- Kernel 3: fp32 row softmax -> bf16 P ---------------------------------
__global__ __launch_bounds__(256) void softmax_kernel(
    const float* __restrict__ S, __bf16* __restrict__ P) {
  size_t row = blockIdx.x;
  const float* s = S + row * TSEQ;
  __bf16* p = P + row * TSEQ;
  int tid = threadIdx.x;
  float v[8];
  float m = -3.0e38f;
#pragma unroll
  for (int i = 0; i < 8; ++i) { v[i] = s[tid + (i << 8)]; m = fmaxf(m, v[i]); }
#pragma unroll
  for (int off = 16; off > 0; off >>= 1) m = fmaxf(m, __shfl_xor(m, off, 32));
  __shared__ float red[8];
  int lane = tid & 31, wid = tid >> 5;
  if (lane == 0) red[wid] = m;
  __syncthreads();
#pragma unroll
  for (int i = 0; i < 8; ++i) m = fmaxf(m, red[i]);
  float sum = 0.0f;
#pragma unroll
  for (int i = 0; i < 8; ++i) { v[i] = __expf(v[i] - m); sum += v[i]; }
#pragma unroll
  for (int off = 16; off > 0; off >>= 1) sum += __shfl_xor(sum, off, 32);
  __syncthreads();
  if (lane == 0) red[wid] = sum;
  __syncthreads();
  float tot = 0.0f;
#pragma unroll
  for (int i = 0; i < 8; ++i) tot += red[i];
  float inv = 1.0f / tot;
#pragma unroll
  for (int i = 0; i < 8; ++i) p[tid + (i << 8)] = (__bf16)(v[i] * inv);
}

// ---- Kernel 4: O = P * V   (B operand = V^T is already [N=c, K=t]) --------
__global__ __launch_bounds__(256) void pv_kernel(
    const __bf16* __restrict__ P, const __bf16* __restrict__ vT,
    __bf16* __restrict__ ao, int b0) {
  int q0 = blockIdx.x * 128;
  int c0 = blockIdx.y * 256;
  int bc = blockIdx.z >> 2, h = blockIdx.z & 3;
  int b = b0 + bc;
  ZERO_ACC
  gemm_core(P + ((size_t)(bc * HEADS + h) * TSEQ + q0) * TSEQ, TSEQ,
            vT + ((size_t)(b * HEADS + h) * CDIM + c0) * TSEQ, TSEQ, TSEQ,
            acc);
  EPILOGUE_COORDS
  const int HC = HEADS * CDIM;  // 2048
  __bf16* out = ao + (size_t)(b * TSEQ + q0) * HC + h * CDIM + c0;
#pragma unroll
  for (int mi = 0; mi < 4; ++mi)
#pragma unroll
    for (int ni = 0; ni < 4; ++ni)
#pragma unroll
      for (int r = 0; r < 8; ++r)
        out[(size_t)(mBase + mi * 16 + r) * HC + nBase + ni * 16] =
            (__bf16)acc[mi][ni][r];
}

// ---- Kernel 5: output projection + bias (fp32 out) ------------------------
__global__ __launch_bounds__(256) void proj_kernel(
    const __bf16* __restrict__ ao, const __bf16* __restrict__ pwb,
    const float* __restrict__ pb, float* __restrict__ out) {
  int m0 = blockIdx.x * 128;
  int n0 = blockIdx.y * 256;
  const int HC = HEADS * CDIM;
  ZERO_ACC
  gemm_core(ao + (size_t)m0 * HC, HC, pwb + (size_t)n0 * HC, HC, HC, acc);
  EPILOGUE_COORDS
#pragma unroll
  for (int mi = 0; mi < 4; ++mi)
#pragma unroll
    for (int ni = 0; ni < 4; ++ni)
#pragma unroll
      for (int r = 0; r < 8; ++r) {
        int n = n0 + nBase + ni * 16;
        out[(size_t)(m0 + mBase + mi * 16 + r) * CDIM + n] =
            acc[mi][ni][r] + pb[n];
      }
}

// ---------------------------------------------------------------------------
extern "C" void kernel_launch(void* const* d_in, const int* in_sizes, int n_in,
                              void* d_out, int out_size, void* d_ws,
                              size_t ws_size, hipStream_t stream) {
  const float* x      = (const float*)d_in[0];
  const float* qkv_w  = (const float*)d_in[1];
  const float* proj_w = (const float*)d_in[2];
  const float* proj_b = (const float*)d_in[3];
  float* out = (float*)d_out;

  const size_t NX = (size_t)BATCH * TSEQ * CDIM;          //  8,388,608
  const size_t NWQ = (size_t)3 * HEADS * CDIM * CDIM;     //  3,145,728
  const size_t NWP = (size_t)CDIM * HEADS * CDIM;         //  1,048,576
  const size_t NE = (size_t)BATCH * HEADS * TSEQ * CDIM;  // 33,554,432
  const size_t SN = (size_t)BCH * HEADS * TSEQ * TSEQ;    // 33,554,432
  __bf16* xb   = (__bf16*)d_ws;
  __bf16* wqb  = xb + NX;
  __bf16* wpb  = wqb + NWQ;
  __bf16* q    = wpb + NWP;
  __bf16* kk   = q + NE;
  __bf16* vT   = kk + NE;
  __bf16* ao   = vT + NE;
  float*  S    = (float*)(ao + NE);
  __bf16* P    = (__bf16*)(S + SN);

  cvt_kernel<<<dim3((unsigned)(NX / 2048)), 256, 0, stream>>>(x, xb);
  cvt_kernel<<<dim3((unsigned)(NWQ / 2048)), 256, 0, stream>>>(qkv_w, wqb);
  cvt_kernel<<<dim3((unsigned)(NWP / 2048)), 256, 0, stream>>>(proj_w, wpb);

  qkv_gemm_kernel<<<dim3(BATCH * TSEQ / 128, 3 * HEADS * CDIM / 256), 256, 0,
                    stream>>>(xb, wqb, q, kk, vT);
  for (int b0 = 0; b0 < BATCH; b0 += BCH) {
    scores_kernel<<<dim3(TSEQ / 128, TSEQ / 256, BCH * HEADS), 256, 0,
                    stream>>>(q, kk, S, b0);
    softmax_kernel<<<dim3((unsigned)(SN / TSEQ)), 256, 0, stream>>>(S, P);
    pv_kernel<<<dim3(TSEQ / 128, CDIM / 256, BCH * HEADS), 256, 0, stream>>>(
        P, vT, ao, b0);
  }
  proj_kernel<<<dim3(BATCH * TSEQ / 128, CDIM / 256), 256, 0, stream>>>(
      ao, proj_w != nullptr ? wpb : wpb, proj_b, out);
}
